// NemotronHTopkRouter_21723944583771
// MI455X (gfx1250) — compile-verified, matrix operand order checked by reading, not run on hardware
//
#include <hip/hip_runtime.h>
#include <hip/hip_bf16.h>

// NemotronH top-k router for MI455X (gfx1250, wave32, WMMA).
// GEMM (8192x256x4096) in bf16 WMMA w/ fp32 accumulate, fused sigmoid +
// group-limited top-k epilogue per 64-token tile.

#define TOKENS  8192
#define HIDDEN  4096
#define NEXP    256
#define NGROUP  8
#define GSIZE   32          // NEXP / NGROUP
#define TOPKG   4
#define TOPK    8
#define SCALING 2.5f

#define BM      64          // tokens per workgroup
#define BK      32          // K per step (one bf16 WMMA depth)
#define KSTEPS  (HIDDEN / BK)
#define LOGIT_LD 257        // padded LDS stride (bank spread)

typedef __attribute__((ext_vector_type(16))) __bf16 bf16x16;
typedef __attribute__((ext_vector_type(8)))  __bf16 bf16x8;
typedef __attribute__((ext_vector_type(8)))  float  f32x8;
typedef __attribute__((ext_vector_type(4)))  float  f32x4;

// Kernel 1: weight fp32 [E][H] -> bf16 K-blocked [H/32][E][32] in workspace.
// Each K-slab of B is then a contiguous 16KB block (perfect L2 streaming).
__global__ void cvt_weight_kernel(const float* __restrict__ w,
                                  __bf16* __restrict__ wb) {
    int idx = blockIdx.x * 256 + threadIdx.x;     // 256*4096 total
    int e = idx >> 12;                            // / HIDDEN
    int k = idx & (HIDDEN - 1);
    wb[(size_t)(k >> 5) * (NEXP * BK) + e * BK + (k & 31)] = (__bf16)w[idx];
}

__global__ __launch_bounds__(256, 2)
void router_kernel(const float* __restrict__ hs,
                   const __bf16* __restrict__ wb,
                   const float* __restrict__ bias,
                   int*   __restrict__ out_idx,
                   float* __restrict__ out_w) {
    __shared__ float sLogit[BM * LOGIT_LD];   // 65792 B
    __shared__ float sBias[NEXP];             // 1024 B

    const int tid  = threadIdx.x;
    const int wave = tid >> 5;
    const int lane = tid & 31;
    const int wm   = wave >> 2;      // 0..1 : which 32-token half
    const int wn   = wave & 3;       // 0..3 : which 64-expert quarter
    const int half = lane >> 4;      // WMMA lane-half (K sub-phase)
    const int r    = lane & 15;      // row/col within 16-wide tile

    sBias[tid] = bias[tid];

    // A rows for this lane's two M-tiles (fp32, converted to bf16 on the fly)
    const float* aRow0 = hs + (size_t)(blockIdx.x * BM + wm * 32 + r) * HIDDEN;
    const float* aRow1 = aRow0 + (size_t)16 * HIDDEN;

    f32x8 acc[2][4];
    #pragma unroll
    for (int mt = 0; mt < 2; ++mt)
        #pragma unroll
        for (int nt = 0; nt < 4; ++nt)
            #pragma unroll
            for (int j = 0; j < 8; ++j)
                acc[mt][nt][j] = 0.0f;

    for (int kb = 0; kb < KSTEPS; ++kb) {
        const int k0 = kb * BK;

        // ---- A fragments: 16 fp32 -> bf16 per lane per M-tile ----
        // v16bf layout: elems 0..7 = K(half*8 .. +7), 8..15 = K(half*8+16 ..)
        bf16x16 a[2];
        #pragma unroll
        for (int mt = 0; mt < 2; ++mt) {
            const float* p = (mt == 0 ? aRow0 : aRow1) + k0 + half * 8;
            f32x4 v0 = *(const f32x4*)(p);
            f32x4 v1 = *(const f32x4*)(p + 4);
            f32x4 v2 = *(const f32x4*)(p + 16);
            f32x4 v3 = *(const f32x4*)(p + 20);
            #pragma unroll
            for (int j = 0; j < 4; ++j) {
                a[mt][j]      = (__bf16)v0[j];
                a[mt][4 + j]  = (__bf16)v1[j];
                a[mt][8 + j]  = (__bf16)v2[j];
                a[mt][12 + j] = (__bf16)v3[j];
            }
        }

        // ---- B fragments straight from L2-resident bf16 blocked weight ----
        const __bf16* wk = wb + (size_t)kb * (NEXP * BK);
        bf16x16 b[4];
        #pragma unroll
        for (int nt = 0; nt < 4; ++nt) {
            const __bf16* q = wk + (wn * 64 + nt * 16 + r) * BK + half * 8;
            bf16x8 lo = *(const bf16x8*)(q);        // K half*8 .. +7
            bf16x8 hi = *(const bf16x8*)(q + 16);   // K half*8+16 .. +23
            b[nt] = __builtin_shufflevector(lo, hi,
                    0,1,2,3,4,5,6,7,8,9,10,11,12,13,14,15);
        }

        // ---- 8 WMMAs: 32 tokens x 64 experts advanced by K=32 ----
        #pragma unroll
        for (int mt = 0; mt < 2; ++mt)
            #pragma unroll
            for (int nt = 0; nt < 4; ++nt)
                acc[mt][nt] = __builtin_amdgcn_wmma_f32_16x16x32_bf16(
                    false, a[mt], false, b[nt],
                    (short)0, acc[mt][nt], false, false);

        // Hint next B slab into cache (global_prefetch_b8)
        if (kb + 1 < KSTEPS)
            __builtin_prefetch(wk + NEXP * BK + lane * 256, 0, 1);
    }

    // ---- C layout: VGPR i -> M = i + half*8, N = r ----
    #pragma unroll
    for (int mt = 0; mt < 2; ++mt)
        #pragma unroll
        for (int nt = 0; nt < 4; ++nt)
            #pragma unroll
            for (int i = 0; i < 8; ++i)
                sLogit[(wm * 32 + mt * 16 + half * 8 + i) * LOGIT_LD
                       + wn * 64 + nt * 16 + r] = acc[mt][nt][i];
    __syncthreads();

    // ---- fused epilogue: scores_for_choice = sigmoid(logit) + bias ----
    #pragma unroll 4
    for (int i = 0; i < (BM * NEXP) / 256; ++i) {  // 64 iters, all 256 thrds
        int idx = i * 256 + tid;
        int t = idx >> 8;
        int e = idx & 255;
        float x = sLogit[t * LOGIT_LD + e];
        float s = 1.0f / (1.0f + __expf(-x));
        sLogit[t * LOGIT_LD + e] = s + sBias[e];
    }
    __syncthreads();

    // ---- per-token group-limited top-k (one thread per token) ----
    if (tid < BM) {
        const float* row = &sLogit[tid * LOGIT_LD];

        // group score = sum of top-2 scores_for_choice within each group
        float gs[NGROUP];
        #pragma unroll
        for (int g = 0; g < NGROUP; ++g) {
            float m1 = -1e30f, m2 = -1e30f;
            for (int j = 0; j < GSIZE; ++j) {
                float v = row[g * GSIZE + j];
                if (v > m1)      { m2 = m1; m1 = v; }
                else if (v > m2) { m2 = v; }
            }
            gs[g] = m1 + m2;
        }

        // top-4 groups (strict > keeps lowest index on ties, like lax.top_k)
        unsigned mask = 0;
        #pragma unroll
        for (int k = 0; k < TOPKG; ++k) {
            float best = -1e30f; int bg = 0;
            #pragma unroll
            for (int g = 0; g < NGROUP; ++g)
                if (gs[g] > best) { best = gs[g]; bg = g; }
            mask |= 1u << bg;
            #pragma unroll
            for (int g = 0; g < NGROUP; ++g)
                if (g == bg) gs[g] = -1e30f;
        }

        // stable top-8 over the 4 selected groups (register insertion sort)
        float bv[TOPK]; int bi[TOPK];
        #pragma unroll
        for (int i = 0; i < TOPK; ++i) { bv[i] = -1e30f; bi[i] = 0; }
        for (int g = 0; g < NGROUP; ++g) {
            if (!((mask >> g) & 1u)) continue;
            for (int j = 0; j < GSIZE; ++j) {
                int e = g * GSIZE + j;
                float v = row[e];
                if (v > bv[TOPK - 1]) {          // strict: equal keeps earlier
                    bv[TOPK - 1] = v; bi[TOPK - 1] = e;
                    #pragma unroll
                    for (int q = TOPK - 1; q > 0; --q) {
                        if (bv[q] > bv[q - 1]) {
                            float tv = bv[q]; bv[q] = bv[q - 1]; bv[q - 1] = tv;
                            int   ti = bi[q]; bi[q] = bi[q - 1]; bi[q - 1] = ti;
                        }
                    }
                }
            }
        }

        // weights = sigmoid scores at selected experts, normalized, * 2.5
        float wv[TOPK]; float wsum = 0.0f;
        #pragma unroll
        for (int i = 0; i < TOPK; ++i) {
            wv[i] = bv[i] - sBias[bi[i]];        // undo bias -> sigmoid score
            wsum += wv[i];
        }
        float scale = SCALING / (wsum + 1e-20f);

        int gt = blockIdx.x * BM + tid;
        #pragma unroll
        for (int i = 0; i < TOPK; ++i) {
            out_idx[gt * TOPK + i] = bi[i];
            out_w[gt * TOPK + i]   = wv[i] * scale;
        }
    }
}

extern "C" void kernel_launch(void* const* d_in, const int* in_sizes, int n_in,
                              void* d_out, int out_size, void* d_ws, size_t ws_size,
                              hipStream_t stream) {
    const float* hs   = (const float*)d_in[0];   // [8192, 4096] fp32
    const float* w    = (const float*)d_in[1];   // [256, 4096] fp32
    const float* bias = (const float*)d_in[2];   // [256] fp32

    __bf16* wb = (__bf16*)d_ws;                  // 2 MB bf16 blocked weight

    int*   out_idx = (int*)d_out;                            // [8192,8] int32
    float* out_w   = (float*)d_out + (size_t)TOKENS * TOPK;  // [8192,8] fp32

    cvt_weight_kernel<<<(NEXP * HIDDEN) / 256, 256, 0, stream>>>(w, wb);
    router_kernel<<<TOKENS / BM, 256, 0, stream>>>(hs, wb, bias, out_idx, out_w);
}
